// TriangularAttention_78228534329663
// MI455X (gfx1250) — matrix operand order repaired and checked by reading, not compiled, not run on hardware
//
#include <hip/hip_runtime.h>

typedef _Float16 half_t;
typedef half_t v16h __attribute__((ext_vector_type(16)));
typedef half_t v8h  __attribute__((ext_vector_type(8)));
typedef half_t v4h  __attribute__((ext_vector_type(4)));
typedef float  v8f  __attribute__((ext_vector_type(8)));
typedef float  v4f  __attribute__((ext_vector_type(4)));
typedef int    v4i  __attribute__((ext_vector_type(4)));
typedef int    v8i  __attribute__((ext_vector_type(8)));
typedef unsigned int v4u __attribute__((ext_vector_type(4)));

#define L_SEQ 256
#define C_DIM 128
#define H_CNT 4
#define D_HEAD 32
#define N_ROWS (L_SEQ * L_SEQ)   // 65536

// ---- CDNA5 async / tensor staging paths, with safe fallbacks --------------
#if defined(__has_builtin)
#if __has_builtin(__builtin_amdgcn_global_load_async_to_lds_b128)
#define HAVE_ASYNC_LDS 1
#endif
#if __has_builtin(__builtin_amdgcn_s_wait_asynccnt)
#define HAVE_WAIT_ASYNC 1
#endif
#if __has_builtin(__builtin_amdgcn_tensor_load_to_lds) && \
    __has_builtin(__builtin_amdgcn_s_wait_tensorcnt)
#define HAVE_TDM 1
#endif
#endif

static __device__ __forceinline__ void cp16(half_t* lds, const half_t* g) {
#if defined(HAVE_ASYNC_LDS)
  __builtin_amdgcn_global_load_async_to_lds_b128(
      (__attribute__((address_space(1))) v4i*)(void*)(g),
      (__attribute__((address_space(3))) v4i*)(void*)(lds), 0, 0);
#else
  *(v8h*)lds = *(const v8h*)g;
#endif
}

static __device__ __forceinline__ void wait_async0() {
#if defined(HAVE_WAIT_ASYNC)
  __builtin_amdgcn_s_wait_asynccnt(0);
#elif defined(HAVE_ASYNC_LDS)
  asm volatile("s_wait_asynccnt 0x0" ::: "memory");
#endif
}

#if defined(HAVE_TDM)
// TDM 2-D tile load: D# per CDNA5 ISA.
// Elements are 2 bytes (data_size code 1). Tile rows land contiguously in LDS.
// This toolchain exposes the 6-arg builtin:
//   (uint32x4 g0, int32x8 g1, int32x4 g2, int32x4 g3, int32x8 g4, i32 cpol)
static __device__ __forceinline__ void tdm_load_2d(
    unsigned int lds_off, const half_t* gptr, unsigned int td0,
    unsigned int td1, unsigned int tile0, unsigned int tile1,
    unsigned long long stride0_elems) {
  unsigned long long ga = (unsigned long long)(uintptr_t)gptr;
  v4u g0;
  g0[0] = 1u;                                  // count=1, user descriptor
  g0[1] = lds_off;                             // lds_addr (bytes)
  g0[2] = (unsigned int)ga;                    // global_addr[31:0]
  g0[3] = (unsigned int)(ga >> 32) | (2u << 30);  // addr[56:32] | type=2
  v8i g1;
  g1[0] = (int)(1u << 16);                     // data_size=1 (2B); mask=0
  g1[1] = (int)((td0 & 0xFFFFu) << 16);        // tensor_dim0[15:0] @ bit48
  g1[2] = (int)(((td0 >> 16) & 0xFFFFu) | ((td1 & 0xFFFFu) << 16));
  g1[3] = (int)(((td1 >> 16) & 0xFFFFu) | ((tile0 & 0xFFFFu) << 16));
  g1[4] = (int)(tile1 & 0xFFFFu);              // tile_dim1; tile_dim2=0 (2-D)
  g1[5] = (int)(unsigned int)(stride0_elems & 0xFFFFFFFFull);
  g1[6] = (int)(unsigned int)((stride0_elems >> 32) & 0xFFFFull);
  g1[7] = 0;                                   // tensor_dim1_stride unused
  v4i gz4 = {};
  v8i gz8 = {};
  __builtin_amdgcn_tensor_load_to_lds(g0, g1, gz4, gz4, gz8, 0);
}
static __device__ __forceinline__ unsigned int lds_offset_of(const void* p) {
  return (unsigned int)(uintptr_t)(__attribute__((address_space(3))) const void*)p;
}
#endif

// ---------------- WMMA helpers (CDNA5 wave32, 16x16x32 f16 -> f32) ----------

static __device__ __forceinline__ v8f wmma_f32_f16(v16h a, v16h b, v8f c) {
  // 8 args: (neg_a, A, neg_b, B, c_mod, C, reuse_a, reuse_b)
  return __builtin_amdgcn_wmma_f32_16x16x32_f16(false, a, false, b, (short)0, c,
                                                false, false);
}

// A-fragment (16x32 f16, row-major source with leading dim `ld`):
// lane m = lane&15 holds row m; halves 0..15 -> K = 16*hi + e,
// where hi = lane>>4.
static __device__ __forceinline__ v16h load_a_frag(const half_t* base, int ld,
                                                   int m, int hi) {
  const half_t* p = base + (size_t)m * ld + (hi << 4);
  v8h lo = *(const v8h*)(p);
  v8h hh = *(const v8h*)(p + 8);
  return __builtin_shufflevector(lo, hh, 0, 1, 2, 3, 4, 5, 6, 7, 8, 9, 10, 11,
                                 12, 13, 14, 15);
}

// Half-wave (16-lane) reductions: rows for lanes 0-15 and 16-31 differ,
// xor masks 1,2,4,8 stay within each 16-lane group.
static __device__ __forceinline__ float half_max(float v) {
  v = fmaxf(v, __shfl_xor(v, 1, 32));
  v = fmaxf(v, __shfl_xor(v, 2, 32));
  v = fmaxf(v, __shfl_xor(v, 4, 32));
  v = fmaxf(v, __shfl_xor(v, 8, 32));
  return v;
}
static __device__ __forceinline__ float half_sum(float v) {
  v += __shfl_xor(v, 1, 32);
  v += __shfl_xor(v, 2, 32);
  v += __shfl_xor(v, 4, 32);
  v += __shfl_xor(v, 8, 32);
  return v;
}
static __device__ __forceinline__ float full_sum(float v) {
  v += __shfl_xor(v, 16, 32);
  v += __shfl_xor(v, 8, 32);
  v += __shfl_xor(v, 4, 32);
  v += __shfl_xor(v, 2, 32);
  v += __shfl_xor(v, 1, 32);
  return v;
}

// ---------------- Kernel 1: convert + transpose weights to f16 -------------
// wt layout: [which][c_out][c_in] f16, which: 0=Wq 1=Wk 2=Wv 3=Wo 4=Wg
__global__ __launch_bounds__(256) void k_wconv(
    const float* __restrict__ Wq, const float* __restrict__ Wk,
    const float* __restrict__ Wv, const float* __restrict__ Wo,
    const float* __restrict__ Wg, half_t* __restrict__ wt) {
  int idx = blockIdx.x * 256 + threadIdx.x;  // 5*16384 = 81920 elements
  int which = idx >> 14;
  int rem = idx & 16383;
  int ci = rem >> 7;   // input channel
  int co = rem & 127;  // output channel
  const float* src = (which == 0)   ? Wq
                     : (which == 1) ? Wk
                     : (which == 2) ? Wv
                     : (which == 3) ? Wo
                                    : Wg;
  wt[(size_t)which * 16384 + (size_t)co * 128 + ci] = (half_t)src[ci * 128 + co];
}

// ---------------- Kernel 2: LayerNorm + f16 casts + bias projection --------
__global__ __launch_bounds__(256) void k_layernorm(
    const float* __restrict__ pair, const float* __restrict__ gamma,
    const float* __restrict__ beta, const float* __restrict__ wbias,
    half_t* __restrict__ z16, half_t* __restrict__ p16,
    float* __restrict__ biasw) {
  int lane = threadIdx.x & 31;
  int wave = threadIdx.x >> 5;
  size_t n = (size_t)blockIdx.x * 8 + wave;  // one wave per row, 65536 rows
  const float* xp = pair + n * C_DIM + lane * 4;
  v4f x = *(const v4f*)xp;
  float s = x[0] + x[1] + x[2] + x[3];
  s = full_sum(s);
  float mu = s * (1.0f / 128.0f);
  float d0 = x[0] - mu, d1 = x[1] - mu, d2 = x[2] - mu, d3 = x[3] - mu;
  float q = d0 * d0 + d1 * d1 + d2 * d2 + d3 * d3;
  q = full_sum(q);
  float rs = rsqrtf(q * (1.0f / 128.0f) + 1e-5f);
  int c0 = lane * 4;
  float y0 = d0 * rs * gamma[c0 + 0] + beta[c0 + 0];
  float y1 = d1 * rs * gamma[c0 + 1] + beta[c0 + 1];
  float y2 = d2 * rs * gamma[c0 + 2] + beta[c0 + 2];
  float y3 = d3 * rs * gamma[c0 + 3] + beta[c0 + 3];
  v4h zh;
  zh[0] = (half_t)y0; zh[1] = (half_t)y1; zh[2] = (half_t)y2; zh[3] = (half_t)y3;
  *(v4h*)(z16 + n * C_DIM + c0) = zh;
  v4h ph;
  ph[0] = (half_t)x[0]; ph[1] = (half_t)x[1];
  ph[2] = (half_t)x[2]; ph[3] = (half_t)x[3];
  *(v4h*)(p16 + n * C_DIM + c0) = ph;
  // bias[h, n] = dot(z_row, Wbias[:, h]); Wbias is [128,4] row-major.
#pragma unroll
  for (int h = 0; h < 4; ++h) {
    float p = y0 * wbias[(c0 + 0) * 4 + h] + y1 * wbias[(c0 + 1) * 4 + h] +
              y2 * wbias[(c0 + 2) * 4 + h] + y3 * wbias[(c0 + 3) * 4 + h];
    p = full_sum(p);
    if (lane == 0) biasw[(size_t)h * N_ROWS + n] = p;
  }
}

// ---------------- Kernel 3: Q/K/V projections (f16 WMMA) -------------------
// Q,K stored row-major [n][c]; V stored pre-transposed per (i,h):
//   vtg[((i*4 + h)*32 + d)*256 + j]
__global__ __launch_bounds__(256) void k_qkv(
    const half_t* __restrict__ z16, const half_t* __restrict__ wt,
    half_t* __restrict__ q16, half_t* __restrict__ k16,
    half_t* __restrict__ vtg) {
  int lane = threadIdx.x & 31;
  int wave = threadIdx.x >> 5;
  int m = lane & 15, hi = lane >> 4;
  int mt = blockIdx.x * 8 + wave;  // 4096 M-tiles of 16 rows
  size_t row0 = (size_t)mt * 16;
  int ib = (int)(row0 >> 8);  // 16-row tiles never cross an i boundary
  v16h a[4];
#pragma unroll
  for (int kk = 0; kk < 4; ++kk)
    a[kk] = load_a_frag(z16 + row0 * C_DIM + kk * 32, C_DIM, m, hi);
#pragma unroll
  for (int w = 0; w < 3; ++w) {
    const half_t* wtp = wt + (size_t)w * 16384;
    for (int nt = 0; nt < 8; ++nt) {
      v8f c = {};
#pragma unroll
      for (int kk = 0; kk < 4; ++kk) {
        v16h b =
            *(const v16h*)(wtp + (size_t)(nt * 16 + m) * C_DIM + kk * 32 + hi * 16);
        c = wmma_f32_f16(a[kk], b, c);
      }
      if (w < 2) {
        // C layout: reg r -> row r + 8*hi, col = m
        half_t* outp = (w == 0) ? q16 : k16;
        half_t* orow = outp + (row0 + hi * 8) * C_DIM + nt * 16;
#pragma unroll
        for (int r = 0; r < 8; ++r) orow[(size_t)r * C_DIM + m] = (half_t)c[r];
      } else {
        // V transposed: col c = nt*16+m -> h = nt>>1, d = (nt&1)*16+m
        int hh = nt >> 1;
        int d = (nt & 1) * 16 + m;
        half_t* vb = vtg + ((size_t)(ib * 4 + hh) * 32 + d) * 256;
        int j0 = (int)(row0 & 255) + 8 * hi;
#pragma unroll
        for (int r = 0; r < 8; ++r) vb[j0 + r] = (half_t)c[r];
      }
    }
  }
}

// ---------------- Kernel 4: flash attention per (i, h) ---------------------
__global__ __launch_bounds__(256) void k_attn(
    const half_t* __restrict__ q16, const half_t* __restrict__ k16,
    const half_t* __restrict__ vtg, const float* __restrict__ biasw,
    half_t* __restrict__ o16) {
  __shared__ __align__(32) half_t Qs[L_SEQ * D_HEAD];   // 16 KB, [j][d]
  __shared__ __align__(32) half_t Ks[L_SEQ * D_HEAD];   // 16 KB, [k][d]
  __shared__ __align__(32) half_t Vts[D_HEAD * L_SEQ];  // 16 KB, [d][k]
  __shared__ __align__(32) half_t Ps[8][16 * 32];       // per-wave P tile
  int i = blockIdx.x >> 2;
  int h = blockIdx.x & 3;
  int tid = threadIdx.x;
#if defined(HAVE_TDM)
  // Stage all three tiles via the Tensor Data Mover: one descriptor each,
  // issued by wave 0 only (TDM ignores EXEC; one issue per instruction).
  if (tid < 32) {
    const half_t* qsrc = q16 + ((size_t)i * L_SEQ) * C_DIM + h * D_HEAD;
    const half_t* ksrc = k16 + ((size_t)i * L_SEQ) * C_DIM + h * D_HEAD;
    const half_t* vsrc = vtg + (size_t)(i * 4 + h) * (D_HEAD * L_SEQ);
    // Q/K: 256 rows x 32 elems, row stride 128 elems -> packed rows in LDS
    tdm_load_2d(lds_offset_of(Qs), qsrc, 32, 256, 32, 256, 128);
    tdm_load_2d(lds_offset_of(Ks), ksrc, 32, 256, 32, 256, 128);
    // V (pre-transposed): one contiguous 8192-element strip
    tdm_load_2d(lds_offset_of(Vts), vsrc, 8192, 1, 8192, 1, 8192);
    __builtin_amdgcn_s_wait_tensorcnt(0);
  }
  __syncthreads();
#else
  {  // async staging: contiguous 64B/thread copies
    size_t base = ((size_t)i * L_SEQ + tid) * C_DIM + h * D_HEAD;
    const half_t* qsrc = q16 + base;
    const half_t* ksrc = k16 + base;
    const half_t* vsrc = vtg + (size_t)(i * 4 + h) * (D_HEAD * L_SEQ) + tid * 32;
    half_t* qdst = Qs + tid * D_HEAD;
    half_t* kdst = Ks + tid * D_HEAD;
    half_t* vdst = Vts + tid * 32;
#pragma unroll
    for (int c = 0; c < 4; ++c) {
      cp16(qdst + c * 8, qsrc + c * 8);
      cp16(kdst + c * 8, ksrc + c * 8);
      cp16(vdst + c * 8, vsrc + c * 8);
    }
  }
  wait_async0();
  __syncthreads();
#endif
  int wave = tid >> 5, lane = tid & 31;
  int m = lane & 15, hi = lane >> 4;
  const float scale = 0.17677669529663687f;  // 1/sqrt(32)
  const float* bbase = biasw + (size_t)h * N_ROWS;
  half_t* ps = Ps[wave];
#pragma unroll 1
  for (int t = 0; t < 2; ++t) {
    int jb = (wave * 2 + t) * 16;  // this wave's j-tile base
    v16h aq = load_a_frag(Qs + jb * D_HEAD, D_HEAD, m, hi);  // full K=32 (=d)
    v8f o0 = {}, o1 = {};
    float mrow[8], lrow[8];
#pragma unroll
    for (int r = 0; r < 8; ++r) { mrow[r] = -1e30f; lrow[r] = 0.0f; }
    const float* brow = bbase + (size_t)(jb + 8 * hi) * L_SEQ;
    for (int ks = 0; ks < L_SEQ; ks += 32) {
      // S tiles: B[d,k] = K[k,d] gathered from row-major Ks (contiguous in d)
      v16h b0 = *(const v16h*)(Ks + (ks + m) * D_HEAD + hi * 16);
      v16h b1 = *(const v16h*)(Ks + (ks + 16 + m) * D_HEAD + hi * 16);
      v8f s0 = {}, s1 = {};
      s0 = wmma_f32_f16(aq, b0, s0);
      s1 = wmma_f32_f16(aq, b1, s1);
#pragma unroll
      for (int r = 0; r < 8; ++r) {
        float bv0 = brow[(size_t)r * L_SEQ + ks + m];
        float bv1 = brow[(size_t)r * L_SEQ + ks + 16 + m];
        float e0 = s0[r] * scale + bv0;
        float e1 = s1[r] * scale + bv1;
        float tm = half_max(fmaxf(e0, e1));
        float nm = fmaxf(mrow[r], tm);
        float corr = __expf(mrow[r] - nm);
        mrow[r] = nm;
        e0 = __expf(e0 - nm);
        e1 = __expf(e1 - nm);
        lrow[r] = lrow[r] * corr + half_sum(e0 + e1);
        o0[r] *= corr;
        o1[r] *= corr;
        // C layout -> row-major P tile [16 x 32] in per-wave LDS scratch
        ps[(r + 8 * hi) * 32 + m] = (half_t)e0;
        ps[(r + 8 * hi) * 32 + 16 + m] = (half_t)e1;
      }
      asm volatile("s_wait_dscnt 0x0" ::: "memory");  // LDS RAW fence (same wave)
      v16h ap = load_a_frag(ps, 32, m, hi);  // P as A fragment (16x32, K=k)
      // V B-fragments from transposed Vts: contiguous in k
      v16h bv0f = *(const v16h*)(Vts + (size_t)m * L_SEQ + ks + hi * 16);
      v16h bv1f = *(const v16h*)(Vts + (size_t)(16 + m) * L_SEQ + ks + hi * 16);
      o0 = wmma_f32_f16(ap, bv0f, o0);
      o1 = wmma_f32_f16(ap, bv1f, o1);
    }
    half_t* orow = o16 + ((size_t)i * L_SEQ + jb + 8 * hi) * C_DIM + h * D_HEAD;
#pragma unroll
    for (int r = 0; r < 8; ++r) {
      float inv = 1.0f / lrow[r];
      orow[(size_t)r * C_DIM + m] = (half_t)(o0[r] * inv);
      orow[(size_t)r * C_DIM + 16 + m] = (half_t)(o1[r] * inv);
    }
  }
}

// ---------------- Kernel 5: output proj + gate + residual ------------------
__global__ __launch_bounds__(256) void k_final(
    const float* __restrict__ pair, const half_t* __restrict__ p16,
    const half_t* __restrict__ o16, const half_t* __restrict__ wt,
    const float* __restrict__ bo, const float* __restrict__ bg,
    float* __restrict__ outp) {
  int lane = threadIdx.x & 31;
  int wave = threadIdx.x >> 5;
  int m = lane & 15, hi = lane >> 4;
  int mt = blockIdx.x * 8 + wave;
  size_t row0 = (size_t)mt * 16;
  const half_t* wto = wt + (size_t)3 * 16384;
  const half_t* wtg = wt + (size_t)4 * 16384;
  v16h ao[4], ag[4];
#pragma unroll
  for (int kk = 0; kk < 4; ++kk) {
    ao[kk] = load_a_frag(o16 + row0 * C_DIM + kk * 32, C_DIM, m, hi);
    ag[kk] = load_a_frag(p16 + row0 * C_DIM + kk * 32, C_DIM, m, hi);
  }
  for (int nt = 0; nt < 8; ++nt) {
    v8f co = {}, cg = {};
#pragma unroll
    for (int kk = 0; kk < 4; ++kk) {
      v16h bo_f =
          *(const v16h*)(wto + (size_t)(nt * 16 + m) * C_DIM + kk * 32 + hi * 16);
      v16h bg_f =
          *(const v16h*)(wtg + (size_t)(nt * 16 + m) * C_DIM + kk * 32 + hi * 16);
      co = wmma_f32_f16(ao[kk], bo_f, co);
      cg = wmma_f32_f16(ag[kk], bg_f, cg);
    }
    int col = nt * 16 + m;
    float bov = bo[col];
    float bgv = bg[col];
    const float* prow = pair + (row0 + 8 * hi) * C_DIM + col;
    float* orow = outp + (row0 + 8 * hi) * C_DIM + col;
#pragma unroll
    for (int r = 0; r < 8; ++r) {
      float gate = 1.0f / (1.0f + __expf(-(cg[r] + bgv)));
      orow[(size_t)r * C_DIM] = prow[(size_t)r * C_DIM] + gate * (co[r] + bov);
    }
  }
}

// ---------------- Launch ---------------------------------------------------
extern "C" void kernel_launch(void* const* d_in, const int* in_sizes, int n_in,
                              void* d_out, int out_size, void* d_ws,
                              size_t ws_size, hipStream_t stream) {
  (void)in_sizes; (void)n_in; (void)out_size; (void)ws_size;
  const float* pair = (const float*)d_in[0];
  const float* ln_g = (const float*)d_in[1];
  const float* ln_b = (const float*)d_in[2];
  const float* Wq = (const float*)d_in[3];
  const float* Wk = (const float*)d_in[4];
  const float* Wv = (const float*)d_in[5];
  const float* Wbias = (const float*)d_in[6];
  const float* Wo = (const float*)d_in[7];
  const float* bo = (const float*)d_in[8];
  const float* Wg = (const float*)d_in[9];
  const float* bg = (const float*)d_in[10];
  float* out = (float*)d_out;

  char* ws = (char*)d_ws;
  const size_t HB = sizeof(half_t);
  const size_t MAT = (size_t)N_ROWS * C_DIM * HB;  // 16 MB each (f16)
  size_t off = 0;
  half_t* wt = (half_t*)(ws + off);    off += (size_t)5 * 16384 * HB;  // 160 KB
  half_t* z16 = (half_t*)(ws + off);   off += MAT;
  half_t* p16 = (half_t*)(ws + off);   off += MAT;
  half_t* q16 = (half_t*)(ws + off);   off += MAT;
  half_t* k16 = (half_t*)(ws + off);   off += MAT;
  half_t* vtg = (half_t*)(ws + off);   off += MAT;  // V pre-transposed
  half_t* o16 = (half_t*)(ws + off);   off += MAT;
  float* biasw = (float*)(ws + off);   off += (size_t)H_CNT * N_ROWS * 4;  // 1 MB

  k_wconv<<<320, 256, 0, stream>>>(Wq, Wk, Wv, Wo, Wg, wt);
  k_layernorm<<<N_ROWS / 8, 256, 0, stream>>>(pair, ln_g, ln_b, Wbias, z16, p16,
                                              biasw);
  k_qkv<<<N_ROWS / 16 / 8, 256, 0, stream>>>(z16, wt, q16, k16, vtg);
  k_attn<<<L_SEQ * H_CNT, 256, 0, stream>>>(q16, k16, vtg, biasw, o16);
  k_final<<<N_ROWS / 16 / 8, 256, 0, stream>>>(pair, p16, o16, wt, bo, bg, out);
}
